// CellLSTM_23338852286892
// MI455X (gfx1250) — matrix operand order repaired and checked by reading, not compiled, not run on hardware
//
#include <hip/hip_runtime.h>

// ---------------------------------------------------------------------------
// LSTM on MI455X (gfx1250): bf16 WMMA (v_wmma_f32_16x16x32_bf16), fp32 accum.
// One kernel launch per timestep = the global barrier for the recurrence.
// x is pre-converted to bf16 for all 512 steps (one-time) so the recurrent
// hot loop contains only b128 loads + WMMA, no fp32->bf16 VALU work.
// ---------------------------------------------------------------------------

typedef __bf16 bf16_t;
typedef __attribute__((ext_vector_type(16))) __bf16 v16bf;
typedef __attribute__((ext_vector_type(8)))  __bf16 v8bf;
typedef __attribute__((ext_vector_type(8)))  float  v8f;

#define SEQ   512
#define BATCH 64
#define EMB   1024
#define HIDN  1024
#define G4    (4 * HIDN)

__device__ __forceinline__ bf16_t f2bf(float f) {
  union { float f; unsigned u; } v; v.f = f;
  unsigned r = v.u + 0x7FFFu + ((v.u >> 16) & 1u);   // round-to-nearest-even
  unsigned short hs = (unsigned short)(r >> 16);
  bf16_t o; __builtin_memcpy(&o, &hs, sizeof(o));
  return o;
}

__device__ __forceinline__ float sigf(float x)      { return 1.0f / (1.0f + __expf(-x)); }
__device__ __forceinline__ float tanhf_fast(float x){ return 2.0f * sigf(2.0f * x) - 1.0f; }

// ---------------------------------------------------------------------------
// Pack: Wx = [W_xi|W_xf|W_ig|W_xo] and Wh = [...] transposed to [n][k] bf16,
// plus concatenated bias. One-time cost; reads are coalesced over n.
// ---------------------------------------------------------------------------
__global__ void pack_weights(const float* __restrict__ Wxi, const float* __restrict__ Wxf,
                             const float* __restrict__ Wig, const float* __restrict__ Wxo,
                             const float* __restrict__ Whi, const float* __restrict__ Whf,
                             const float* __restrict__ Whg, const float* __restrict__ Who,
                             const float* __restrict__ bi,  const float* __restrict__ bf_,
                             const float* __restrict__ bg,  const float* __restrict__ bo,
                             bf16_t* __restrict__ WxT, bf16_t* __restrict__ WhT,
                             float* __restrict__ bias)
{
  int idx = blockIdx.x * blockDim.x + threadIdx.x;        // over 4096*1024
  if (idx < G4 * EMB) {
    int k    = idx >> 12;          // 0..1023
    int n    = idx & (G4 - 1);     // 0..4095
    int gate = n >> 10;
    int col  = n & (HIDN - 1);
    const float* Wx = (gate == 0) ? Wxi : (gate == 1) ? Wxf : (gate == 2) ? Wig : Wxo;
    const float* Wh = (gate == 0) ? Whi : (gate == 1) ? Whf : (gate == 2) ? Whg : Who;
    WxT[(size_t)n * EMB  + k] = f2bf(Wx[(size_t)k * HIDN + col]);
    WhT[(size_t)n * HIDN + k] = f2bf(Wh[(size_t)k * HIDN + col]);
  }
  if (idx < G4) {
    int gate = idx >> 10, col = idx & (HIDN - 1);
    const float* bsrc = (gate == 0) ? bi : (gate == 1) ? bf_ : (gate == 2) ? bg : bo;
    bias[idx] = bsrc[col];
  }
}

// One-time: convert all SEQ steps of x to bf16 (keeps recurrent loop VALU-free).
__global__ void pack_x(const float* __restrict__ x, bf16_t* __restrict__ xbf, long long n4)
{
  long long i = (long long)blockIdx.x * blockDim.x + threadIdx.x;   // over n4 float4s
  if (i < n4) {
    float4 f = ((const float4*)x)[i];
    bf16_t* o = xbf + i * 4;
    o[0] = f2bf(f.x); o[1] = f2bf(f.y); o[2] = f2bf(f.z); o[3] = f2bf(f.w);
  }
}

__global__ void init_state(const float* __restrict__ h0, const float* __restrict__ c0,
                           bf16_t* __restrict__ hbuf, float* __restrict__ cbuf)
{
  int idx = blockIdx.x * blockDim.x + threadIdx.x;
  if (idx < BATCH * HIDN) {
    hbuf[idx] = f2bf(h0[idx]);
    cbuf[idx] = c0[idx];
  }
}

// ---------------------------------------------------------------------------
// A-tile loader (16x32, this file's k order, same as B): lane m = lane&15; 16 contiguous
// K elements at k0 + 16*(lane>>4).  src is row-major bf16, row stride 1024.
// ---------------------------------------------------------------------------
__device__ __forceinline__ v16bf load_a_bf(const bf16_t* __restrict__ base,
                                           int row, int k0, int nl, int g16) {
  const bf16_t* p = base + (size_t)row * 1024 + k0 + g16 * 16;
  v8bf lo = *(const v8bf*)(p);
  v8bf hi = *(const v8bf*)(p + 8);
  return __builtin_shufflevector(lo, hi, 0, 1, 2, 3, 4, 5, 6, 7,
                                         8, 9, 10, 11, 12, 13, 14, 15);
}

// ---------------------------------------------------------------------------
// One LSTM timestep (fast path: x already bf16). Grid = 128 single-wave blocks.
//   block b: hidden-column slice colb = (b & 63)*16 (same slice in all 4 gates)
//            batch-row half    mrow0 = (b >> 6)*32
// Each wave: 8 fp32 accum tiles (4 gates x 2 row-tiles), K = 2048.
// ---------------------------------------------------------------------------
__global__ __launch_bounds__(32) void lstm_step_fast(
    const bf16_t* __restrict__ xbf,    // [BATCH][EMB] bf16 (this step)
    const bf16_t* __restrict__ WxT,    // [4H][EMB]  transposed bf16
    const bf16_t* __restrict__ WhT,    // [4H][HIDN] transposed bf16
    const float*  __restrict__ bias,   // [4H]
    const bf16_t* __restrict__ h_in,   // [BATCH][HIDN]
    bf16_t*       __restrict__ h_out,  // [BATCH][HIDN]
    float*        __restrict__ c_ws,   // [BATCH][HIDN]
    float*        __restrict__ out)    // [BATCH][HIDN] (fp32 result)
{
  const int lane  = threadIdx.x;
  const int nl    = lane & 15;
  const int g16   = lane >> 4;
  const int colb  = (blockIdx.x & 63) * 16;
  const int col   = colb + nl;
  const int mrow0 = (blockIdx.x >> 6) * 32;

  v8f acc[4][2];
#pragma unroll
  for (int g = 0; g < 4; ++g)
#pragma unroll
    for (int mt = 0; mt < 2; ++mt)
      acc[g][mt] = (v8f){0.f, 0.f, 0.f, 0.f, 0.f, 0.f, 0.f, 0.f};

  // ---- Phase 1: x_t @ Wx (pure bf16 loads + WMMA) -------------------------
#pragma unroll 2
  for (int k0 = 0; k0 < EMB; k0 += 32) {
    v16bf A0 = load_a_bf(xbf, mrow0 + nl,      k0, nl, g16);
    v16bf A1 = load_a_bf(xbf, mrow0 + 16 + nl, k0, nl, g16);
#pragma unroll
    for (int g = 0; g < 4; ++g) {
      const bf16_t* wp = WxT + (size_t)(g * HIDN + colb + nl) * EMB + k0 + g16 * 16;
      v16bf b = *(const v16bf*)wp;
      acc[g][0] = __builtin_amdgcn_wmma_f32_16x16x32_bf16(false, A0, false, b,
                                                          (short)0, acc[g][0], false, false);
      acc[g][1] = __builtin_amdgcn_wmma_f32_16x16x32_bf16(false, A1, false, b,
                                                          (short)0, acc[g][1], false, false);
    }
  }

  // ---- Phase 2: h_{t-1} @ Wh ----------------------------------------------
#pragma unroll 2
  for (int k0 = 0; k0 < HIDN; k0 += 32) {
    v16bf A0 = load_a_bf(h_in, mrow0 + nl,      k0, nl, g16);
    v16bf A1 = load_a_bf(h_in, mrow0 + 16 + nl, k0, nl, g16);
#pragma unroll
    for (int g = 0; g < 4; ++g) {
      const bf16_t* wp = WhT + (size_t)(g * HIDN + colb + nl) * HIDN + k0 + g16 * 16;
      v16bf b = *(const v16bf*)wp;
      acc[g][0] = __builtin_amdgcn_wmma_f32_16x16x32_bf16(false, A0, false, b,
                                                          (short)0, acc[g][0], false, false);
      acc[g][1] = __builtin_amdgcn_wmma_f32_16x16x32_bf16(false, A1, false, b,
                                                          (short)0, acc[g][1], false, false);
    }
  }

  // ---- Gates + state update (slice-local, fp32) ---------------------------
  const float bi_ = bias[col];
  const float bf2 = bias[HIDN + col];
  const float bg_ = bias[2 * HIDN + col];
  const float bo_ = bias[3 * HIDN + col];

#pragma unroll
  for (int mt = 0; mt < 2; ++mt) {
#pragma unroll
    for (int r = 0; r < 8; ++r) {
      const int row = mrow0 + mt * 16 + g16 * 8 + r;   // C layout: M = r + 8*laneHalf
      const size_t idx = (size_t)row * HIDN + col;
      const float zi = acc[0][mt][r] + bi_;
      const float zf = acc[1][mt][r] + bf2;
      const float zg = acc[2][mt][r] + bg_;
      const float zo = acc[3][mt][r] + bo_;
      const float ig = sigf(zi);
      const float fg = sigf(zf);
      const float gg = tanhf_fast(zg);
      const float og = sigf(zo);
      const float cn = fg * c_ws[idx] + ig * gg;
      c_ws[idx] = cn;
      const float hv = og * tanhf_fast(cn);
      out[idx]   = hv;
      h_out[idx] = f2bf(hv);
    }
  }
}

// ---------------------------------------------------------------------------
// Fallback timestep (workspace too small to cache bf16 x): convert x on the fly.
// ---------------------------------------------------------------------------
__global__ __launch_bounds__(32) void lstm_step_cvt(
    const float*  __restrict__ x,
    const bf16_t* __restrict__ WxT,
    const bf16_t* __restrict__ WhT,
    const float*  __restrict__ bias,
    const bf16_t* __restrict__ h_in,
    bf16_t*       __restrict__ h_out,
    float*        __restrict__ c_ws,
    float*        __restrict__ out)
{
  const int lane  = threadIdx.x;
  const int nl    = lane & 15;
  const int g16   = lane >> 4;
  const int colb  = (blockIdx.x & 63) * 16;
  const int col   = colb + nl;
  const int mrow0 = (blockIdx.x >> 6) * 32;

  v8f acc[4][2];
#pragma unroll
  for (int g = 0; g < 4; ++g)
#pragma unroll
    for (int mt = 0; mt < 2; ++mt)
      acc[g][mt] = (v8f){0.f, 0.f, 0.f, 0.f, 0.f, 0.f, 0.f, 0.f};

#pragma unroll 2
  for (int k0 = 0; k0 < EMB; k0 += 32) {
    v16bf A[2];
#pragma unroll
    for (int mt = 0; mt < 2; ++mt) {
      const float* p = x + (size_t)(mrow0 + mt * 16 + nl) * EMB + k0 + g16 * 16;
      float4 fa = *(const float4*)(p);
      float4 fb = *(const float4*)(p + 4);
      float4 fc = *(const float4*)(p + 8);
      float4 fd = *(const float4*)(p + 12);
      v16bf a;
      a[0]  = f2bf(fa.x); a[1]  = f2bf(fa.y); a[2]  = f2bf(fa.z); a[3]  = f2bf(fa.w);
      a[4]  = f2bf(fb.x); a[5]  = f2bf(fb.y); a[6]  = f2bf(fb.z); a[7]  = f2bf(fb.w);
      a[8]  = f2bf(fc.x); a[9]  = f2bf(fc.y); a[10] = f2bf(fc.z); a[11] = f2bf(fc.w);
      a[12] = f2bf(fd.x); a[13] = f2bf(fd.y); a[14] = f2bf(fd.z); a[15] = f2bf(fd.w);
      A[mt] = a;
    }
#pragma unroll
    for (int g = 0; g < 4; ++g) {
      const bf16_t* wp = WxT + (size_t)(g * HIDN + colb + nl) * EMB + k0 + g16 * 16;
      v16bf b = *(const v16bf*)wp;
      acc[g][0] = __builtin_amdgcn_wmma_f32_16x16x32_bf16(false, A[0], false, b,
                                                          (short)0, acc[g][0], false, false);
      acc[g][1] = __builtin_amdgcn_wmma_f32_16x16x32_bf16(false, A[1], false, b,
                                                          (short)0, acc[g][1], false, false);
    }
  }

#pragma unroll 2
  for (int k0 = 0; k0 < HIDN; k0 += 32) {
    v16bf A0 = load_a_bf(h_in, mrow0 + nl,      k0, nl, g16);
    v16bf A1 = load_a_bf(h_in, mrow0 + 16 + nl, k0, nl, g16);
#pragma unroll
    for (int g = 0; g < 4; ++g) {
      const bf16_t* wp = WhT + (size_t)(g * HIDN + colb + nl) * HIDN + k0 + g16 * 16;
      v16bf b = *(const v16bf*)wp;
      acc[g][0] = __builtin_amdgcn_wmma_f32_16x16x32_bf16(false, A0, false, b,
                                                          (short)0, acc[g][0], false, false);
      acc[g][1] = __builtin_amdgcn_wmma_f32_16x16x32_bf16(false, A1, false, b,
                                                          (short)0, acc[g][1], false, false);
    }
  }

  const float bi_ = bias[col];
  const float bf2 = bias[HIDN + col];
  const float bg_ = bias[2 * HIDN + col];
  const float bo_ = bias[3 * HIDN + col];

#pragma unroll
  for (int mt = 0; mt < 2; ++mt) {
#pragma unroll
    for (int r = 0; r < 8; ++r) {
      const int row = mrow0 + mt * 16 + g16 * 8 + r;
      const size_t idx = (size_t)row * HIDN + col;
      const float zi = acc[0][mt][r] + bi_;
      const float zf = acc[1][mt][r] + bf2;
      const float zg = acc[2][mt][r] + bg_;
      const float zo = acc[3][mt][r] + bo_;
      const float ig = sigf(zi);
      const float fg = sigf(zf);
      const float gg = tanhf_fast(zg);
      const float og = sigf(zo);
      const float cn = fg * c_ws[idx] + ig * gg;
      c_ws[idx] = cn;
      const float hv = og * tanhf_fast(cn);
      out[idx]   = hv;
      h_out[idx] = f2bf(hv);
    }
  }
}

// ---------------------------------------------------------------------------
// Host-side launcher. Workspace layout (bytes):
//   [0,  8M)      WxT  bf16 [4096][1024]
//   [8M, 16M)     WhT  bf16 [4096][1024]
//   [16M, +16KB)  bias fp32 [4096]
//   h ping / h pong (bf16 [64][1024]), c fp32 [64][1024]      (~16.65 MB)
//   xbf bf16 [512][64][1024] (64 MB)  -- only if ws_size permits (~80.7 MB)
// ---------------------------------------------------------------------------
extern "C" void kernel_launch(void* const* d_in, const int* in_sizes, int n_in,
                              void* d_out, int out_size, void* d_ws, size_t ws_size,
                              hipStream_t stream) {
  (void)in_sizes; (void)n_in; (void)out_size;

  const float* inputs = (const float*)d_in[0];
  const float* Wxi = (const float*)d_in[1];
  const float* Wxf = (const float*)d_in[2];
  const float* Wig = (const float*)d_in[3];
  const float* Wxo = (const float*)d_in[4];
  const float* Whi = (const float*)d_in[5];
  const float* Whf = (const float*)d_in[6];
  const float* Whg = (const float*)d_in[7];
  const float* Who = (const float*)d_in[8];
  const float* bi  = (const float*)d_in[9];
  const float* bf_ = (const float*)d_in[10];
  const float* bg  = (const float*)d_in[11];
  const float* bo  = (const float*)d_in[12];
  const float* h0  = (const float*)d_in[13];
  const float* c0  = (const float*)d_in[14];

  char* ws = (char*)d_ws;
  const size_t szW    = (size_t)G4 * EMB * sizeof(bf16_t);          // 8 MB each
  const size_t szH    = (size_t)BATCH * HIDN * sizeof(bf16_t);      // 128 KB
  const size_t szC    = (size_t)BATCH * HIDN * sizeof(float);       // 256 KB
  const size_t szX    = (size_t)SEQ * BATCH * EMB * sizeof(bf16_t); // 64 MB

  bf16_t* WxT  = (bf16_t*)(ws);
  bf16_t* WhT  = (bf16_t*)(ws + szW);
  float*  bias = (float*) (ws + 2 * szW);
  char*   p    = ws + 2 * szW + G4 * sizeof(float);
  bf16_t* hA   = (bf16_t*)(p);
  bf16_t* hB   = (bf16_t*)(p + szH);
  float*  cbuf = (float*) (p + 2 * szH);
  char*   pend = p + 2 * szH + szC;
  bf16_t* xbf  = (bf16_t*)(pend);
  const bool useXCache = ws_size >= (size_t)(pend - ws) + szX;

  // One-time packs + state init
  pack_weights<<<(G4 * EMB + 255) / 256, 256, 0, stream>>>(
      Wxi, Wxf, Wig, Wxo, Whi, Whf, Whg, Who, bi, bf_, bg, bo, WxT, WhT, bias);
  init_state<<<(BATCH * HIDN + 255) / 256, 256, 0, stream>>>(h0, c0, hA, cbuf);
  if (useXCache) {
    const long long n4 = (long long)SEQ * BATCH * EMB / 4;
    pack_x<<<(unsigned)((n4 + 255) / 256), 256, 0, stream>>>(inputs, xbf, n4);
  }

  float* out = (float*)d_out;
  for (int t = 0; t < SEQ; ++t) {
    bf16_t* hin  = (t & 1) ? hB : hA;
    bf16_t* hout = (t & 1) ? hA : hB;
    if (useXCache) {
      lstm_step_fast<<<128, 32, 0, stream>>>(
          xbf + (size_t)t * BATCH * EMB, WxT, WhT, bias,
          hin, hout, cbuf, out + (size_t)t * BATCH * HIDN);
    } else {
      lstm_step_cvt<<<128, 32, 0, stream>>>(
          inputs + (size_t)t * BATCH * EMB, WxT, WhT, bias,
          hin, hout, cbuf, out + (size_t)t * BATCH * HIDN);
    }
  }
}
